// GaussianModel2D_82875688944067
// MI455X (gfx1250) — compile-verified, matrix operand order checked by reading, not run on hardware
//
#include <hip/hip_runtime.h>

typedef __attribute__((ext_vector_type(16))) _Float16 v16h;
typedef __attribute__((ext_vector_type(8)))  float    v8f;

#define IMG_H     512
#define IMG_W     512
#define NGAUSS    16384
#define CHUNK_G   256          // gaussians staged in LDS per block iteration
#define THREADS   256          // 8 waves (wave32)
#define WAVES     (THREADS / 32)
#define TILES     4            // WMMA accumulator tiles (16 pixels each) per wave
// 8 waves * 4 tiles * 16 px = 512 px = exactly one image row per block
#define PIX_PER_BLOCK (WAVES * TILES * 16)

#define HALF_LOG2E 0.72134752f               // 0.5 * log2(e)
#define INV_STEP   (1.0f / 511.0f)           // linspace(0,1,512) step

__global__ __launch_bounds__(THREADS)
void gauss2d_splat_wmma(const float* __restrict__ positions,     // [N,2] (y,x)
                        const float* __restrict__ log_scales,    // [N,2]
                        const float* __restrict__ rotations,     // [N]
                        const float* __restrict__ color_logits,  // [N,3]
                        const float* __restrict__ opacity_logits,// [N]
                        float* __restrict__ out)                 // [H,W,3] f32
{
    // Per-gaussian, per-row folded quadratic:  e(gx) = qA*gx^2 + B2*gx + C2
    __shared__ float4 s_geo[CHUNK_G];   // (qA, B2, C2, 0)
    __shared__ float4 s_col[CHUNK_G];   // (a*r, a*g, a*b, 0)

    const int tid  = threadIdx.x;
    const int lane = tid & 31;
    const int wave = tid >> 5;
    const int ncol = lane & 15;        // B/D column (channel); A row (pixel-in-tile)
    const int khi  = (lane >> 4) & 1;  // lane half: +8 on K (A/B) and on M (D rows)

    const int   row = blockIdx.x;            // this block renders image row `row`
    const float gy  = (float)row * INV_STEP; // block-uniform grid_y

    // grid_x for this lane's A-matrix row in each tile
    float gxs[TILES];
    int   x0[TILES];
#pragma unroll
    for (int t = 0; t < TILES; ++t) {
        x0[t]  = (wave * TILES + t) * 16;
        gxs[t] = (float)(x0[t] + ncol) * INV_STEP;
    }

    v8f acc[TILES];
#pragma unroll
    for (int t = 0; t < TILES; ++t)
#pragma unroll
        for (int r = 0; r < 8; ++r) acc[t][r] = 0.0f;

    for (int cb = 0; cb < NGAUSS; cb += CHUNK_G) {
        __syncthreads();   // previous iteration's readers are done

        // ---- stage: one gaussian per thread, fold row geometry into coeffs ----
        {
            const int g = cb + tid;
            float py = positions[2 * g + 0];
            float px = positions[2 * g + 1];
            float sx = __expf(log_scales[2 * g + 0]);
            float sy = __expf(log_scales[2 * g + 1]);
            sx = fminf(fmaxf(sx, 1e-5f), 1.0f);
            sy = fminf(fmaxf(sy, 1e-5f), 1.0f);
            float cx2 = HALF_LOG2E / (sx * sx);   // exp->exp2 and -0.5 folded in
            float cy2 = HALF_LOG2E / (sy * sy);
            float th = rotations[g];
            float ct = __cosf(th);
            float st = __sinf(th);
            // quadratic form in (dx,dy)
            float qA = cx2 * ct * ct + cy2 * st * st;
            float qB = 2.0f * ct * st * (cx2 - cy2);
            float qC = cx2 * st * st + cy2 * ct * ct;
            // fold dy (block-uniform row) -> quadratic in dx
            float dy    = gy - py;
            float Bline = qB * dy;
            float Cline = qC * dy * dy;
            // expand dx = gx - px -> quadratic in gx
            float B2 = Bline - 2.0f * qA * px;
            float C2 = qA * px * px - Bline * px + Cline;
            s_geo[tid] = make_float4(qA, B2, C2, 0.0f);

            float al = 1.0f / (1.0f + __expf(-opacity_logits[g]));
            float cr = 1.0f / (1.0f + __expf(-color_logits[3 * g + 0]));
            float cg = 1.0f / (1.0f + __expf(-color_logits[3 * g + 1]));
            float cbl= 1.0f / (1.0f + __expf(-color_logits[3 * g + 2]));
            s_col[tid] = make_float4(al * cr, al * cg, al * cbl, 0.0f);
        }
        __syncthreads();

        // ---- 8 K-steps of 32 gaussians ----
        for (int k0 = 0; k0 < CHUNK_G; k0 += 32) {
            // One b128 load per gaussian; reused across all 4 tiles.
            float4 gp[16];
#pragma unroll
            for (int j = 0; j < 16; ++j) {
                int k = k0 + (j < 8 ? j : j + 8) + khi * 8;
                gp[j] = s_geo[k];
            }

            // B operand (32x16 f16): column = channel, only cols 0..2 nonzero.
            v16h Bm;
#pragma unroll
            for (int j = 0; j < 16; ++j) {
                int k = k0 + (j < 8 ? j : j + 8) + khi * 8;
                float v = (ncol < 3) ? ((const float*)&s_col[k])[ncol] : 0.0f;
                Bm[j] = (_Float16)v;
            }

#pragma unroll
            for (int t = 0; t < TILES; ++t) {
                const float gx = gxs[t];
                v16h Am;
#pragma unroll
                for (int j = 0; j < 16; ++j) {
                    // e = qA*gx^2 + B2*gx + C2 : exactly 2 FMAs per pair
                    float m = __builtin_fmaf(gp[j].x, gx, gp[j].y);
                    float e = __builtin_fmaf(m, gx, gp[j].z);
                    Am[j] = (_Float16)__builtin_amdgcn_exp2f(-e);
                }
                acc[t] = __builtin_amdgcn_wmma_f32_16x16x32_f16(
                    /*neg_a=*/false, Am, /*neg_b=*/false, Bm,
                    /*c_mod=*/(short)0, acc[t],
                    /*reuse_a=*/false, /*reuse_b=*/false);
            }
        }
    }

    // ---- store: D lanes 0-15 = channels, VGPR r = pixel row M ----
    if (ncol < 3) {
#pragma unroll
        for (int t = 0; t < TILES; ++t) {
#pragma unroll
            for (int r = 0; r < 8; ++r) {
                int x = x0[t] + r + khi * 8;         // lanes>=16 hold M=8..15
                float v = fminf(fmaxf(acc[t][r], 0.0f), 1.0f);
                out[(row * IMG_W + x) * 3 + ncol] = v;
            }
        }
    }
}

extern "C" void kernel_launch(void* const* d_in, const int* in_sizes, int n_in,
                              void* d_out, int out_size, void* d_ws, size_t ws_size,
                              hipStream_t stream) {
    (void)in_sizes; (void)n_in; (void)d_ws; (void)ws_size; (void)out_size;
    const float* positions      = (const float*)d_in[0];
    const float* log_scales     = (const float*)d_in[1];
    const float* rotations      = (const float*)d_in[2];
    const float* color_logits   = (const float*)d_in[3];
    const float* opacity_logits = (const float*)d_in[4];
    float* out = (float*)d_out;

    dim3 grid(IMG_H);        // one block per image row (512 blocks)
    dim3 block(THREADS);     // 8 wave32 waves
    gauss2d_splat_wmma<<<grid, block, 0, stream>>>(
        positions, log_scales, rotations, color_logits, opacity_logits, out);
}
